// SelfAttention_70918499992238
// MI455X (gfx1250) — matrix operand order repaired and checked by reading, compiled, not run on hardware
//
#include <hip/hip_runtime.h>
#include <hip/hip_bf16.h>

// ---------------------------------------------------------------------------
// Causal self-attention forward, CDNA5 (gfx1250), bf16 WMMA pipeline.
//   B=4, S=2048, E=K=V=1024
// v3: bank-conflict-free LDS strides (64 banks x 4B -> pad rows so each
//     wave32 ds_load_b128 hits the 2-pass hardware floor), on top of v2's
//     async global->LDS double-buffered GEMM and LDS-staged attention.
// ---------------------------------------------------------------------------

#define BATCH 4
#define SEQ   2048
#define EMB   1024   // = DIM_K = DIM_V

typedef __attribute__((ext_vector_type(16))) __bf16       bf16x16;
typedef __attribute__((ext_vector_type(8)))  float        f32x8;
typedef __attribute__((ext_vector_type(4)))  unsigned int u32x4;

struct Frag32 { u32x4 a, b; };   // 32 bytes == one bf16x16 fragment

// round-to-nearest-even f32 -> bf16
static __device__ inline unsigned short f2bf(float f) {
    unsigned int u = __builtin_bit_cast(unsigned int, f);
    u += 0x7FFFu + ((u >> 16) & 1u);
    return (unsigned short)(u >> 16);
}

// generic pointer to LDS byte address (low 32 bits of shared-aperture addr)
static __device__ inline unsigned lds_off(const void* p) {
    return (unsigned)(unsigned long long)p;
}

// CDNA5 async copy: 16B global -> LDS, tracked by ASYNCcnt (no VGPR transit)
static __device__ inline void async_g2l_b128(unsigned laddr,
                                             const unsigned short* g) {
    asm volatile("global_load_async_to_lds_b128 %0, %1, off"
                 :: "v"(laddr), "v"(g) : "memory");
}
static __device__ inline void wait_async_0() {
    asm volatile("s_wait_asynccnt 0" ::: "memory");
}
static __device__ inline void wait_async_4() {   // retire current stage only
    asm volatile("s_wait_asynccnt 4" ::: "memory");
}

// A fragment (16x32, bf16): lane m, 16 contiguous K values at p
static __device__ inline bf16x16 load_a_frag(const unsigned short* p) {
    Frag32 t;
    t.a = *(const u32x4*)(p);
    t.b = *(const u32x4*)(p + 8);
    return __builtin_bit_cast(bf16x16, t);
}

// B fragment (32x16, bf16): lane n, 16 contiguous K values at p
static __device__ inline bf16x16 load_b_frag(const unsigned short* p) {
    Frag32 t;
    t.a = *(const u32x4*)(p);
    t.b = *(const u32x4*)(p + 8);
    return __builtin_bit_cast(bf16x16, t);
}

static __device__ inline f32x8 wmma_bf16(bf16x16 a, bf16x16 b, f32x8 c) {
    return __builtin_amdgcn_wmma_f32_16x16x32_bf16(
        /*neg_a=*/false, a, /*neg_b=*/false, b,
        /*c_mod=*/(short)0, c, /*reuse_a=*/false, /*reuse_b=*/false);
}

// ---------------------------------------------------------------------------
// f32 -> bf16 elementwise convert (row-major copy)
// ---------------------------------------------------------------------------
__global__ __launch_bounds__(256)
void convert_bf16_kernel(const float* __restrict__ in,
                         unsigned short* __restrict__ out, int n) {
    int i = blockIdx.x * 256 + threadIdx.x;
    if (i < n) out[i] = f2bf(in[i]);
}

// ---------------------------------------------------------------------------
// f32 W[E][N] -> bf16 Wt[N][E] (transpose so GEMM B-operand is K-contiguous)
// ---------------------------------------------------------------------------
__global__ __launch_bounds__(256)
void convert_wT_kernel(const float* __restrict__ W,
                       unsigned short* __restrict__ Wt, int E, int N) {
    int i = blockIdx.x * 256 + threadIdx.x;
    if (i >= E * N) return;
    int e = i / N, n = i % N;
    Wt[(long)n * E + e] = f2bf(W[i]);
}

// ---------------------------------------------------------------------------
// GEMM: C[M x N] = A[M x Kd] * Bt[N x Kd]^T      (bf16 in, f32 accumulate)
//   mode 0: bf16 row-major        (Q, K projections)
//   mode 1: bf16 transposed/batch (V projection -> Vt[b][n][s])
//   mode 2: f32 row-major         (final output projection)
// Block: 256 thr (8 waves), C tile 128x128 (waves 2x4, 64x32 each).
// k-slices of 32 staged to LDS via async copies, double buffered.
// LDS tile row stride padded 32->40 elems (80B = 20 banks): conflict-free.
// ---------------------------------------------------------------------------
#define TKP 40   // padded LDS row stride (elements) for 32-deep k slices

__global__ __launch_bounds__(256)
void gemm_bf16_kernel(const unsigned short* __restrict__ A,
                      const unsigned short* __restrict__ Bt,
                      void* __restrict__ Cout,
                      int M, int N, int Kd, int mode) {
    __shared__ unsigned short Asm[2][128 * TKP];  // 10KB per buffer
    __shared__ unsigned short Bsm[2][128 * TKP];

    const int tid  = threadIdx.x;
    const int lane = tid & 31;
    const int wave = tid >> 5;
    const int l15  = lane & 15;
    const bool hiL = lane >= 16;

    const int nTilesN = N >> 7;
    const int row0 = (blockIdx.x / nTilesN) << 7;
    const int col0 = (blockIdx.x % nTilesN) << 7;
    const int wm = (wave >> 2) * 64;              // wave row offset in tile
    const int wn = (wave & 3) * 32;               // wave col offset in tile

    // fill mapping: 2 threads per 32-element row; each issues 2 x b128 (A,B)
    const int fr = tid >> 1;                      // row 0..127
    const int fo = (tid & 1) << 4;                // element 0 or 16
    const unsigned short* gA = A  + (long)(row0 + fr) * Kd + fo;
    const unsigned short* gB = Bt + (long)(col0 + fr) * Kd + fo;
    const unsigned lA = lds_off(&Asm[0][0]) + (unsigned)(fr * TKP + fo) * 2;
    const unsigned lB = lds_off(&Bsm[0][0]) + (unsigned)(fr * TKP + fo) * 2;

    auto fill = [&](int buf, int k0) {            // 4 async b128 per thread
        const unsigned bo = (unsigned)buf * (128 * TKP * 2);
        async_g2l_b128(lA + bo,      gA + k0);
        async_g2l_b128(lA + bo + 16, gA + k0 + 8);
        async_g2l_b128(lB + bo,      gB + k0);
        async_g2l_b128(lB + bo + 16, gB + k0 + 8);
    };

    f32x8 acc[4][2];
    const f32x8 zero = {0.f,0.f,0.f,0.f,0.f,0.f,0.f,0.f};
#pragma unroll
    for (int r = 0; r < 4; ++r)
#pragma unroll
        for (int c = 0; c < 2; ++c) acc[r][c] = zero;

    const int nK = Kd >> 5;
    fill(0, 0);
    for (int kt = 0; kt < nK; ++kt) {
        const int buf = kt & 1;
        if (kt + 1 < nK) {                        // overlap next fill w/ math
            fill(buf ^ 1, (kt + 1) << 5);
            wait_async_4();                       // only current 4 must land
        } else {
            wait_async_0();
        }
        __syncthreads();

        const unsigned short* As = &Asm[buf][0];
        const unsigned short* Bs = &Bsm[buf][0];
        bf16x16 aF[4], bF[2];
#pragma unroll
        for (int r = 0; r < 4; ++r)
            aF[r] = load_a_frag(As + (wm + r * 16 + l15) * TKP + (hiL ? 16 : 0));
#pragma unroll
        for (int c = 0; c < 2; ++c)
            bF[c] = load_b_frag(Bs + (wn + c * 16 + l15) * TKP + (hiL ? 16 : 0));
#pragma unroll
        for (int r = 0; r < 4; ++r)
#pragma unroll
            for (int c = 0; c < 2; ++c)
                acc[r][c] = wmma_bf16(aF[r], bF[c], acc[r][c]);
        __syncthreads();
    }

#pragma unroll
    for (int r = 0; r < 4; ++r)
#pragma unroll
        for (int c = 0; c < 2; ++c)
#pragma unroll
            for (int rr = 0; rr < 8; ++rr) {
                int row = row0 + wm + r * 16 + rr + (hiL ? 8 : 0);
                int col = col0 + wn + c * 16 + l15;
                float v = acc[r][c][rr];
                if (mode == 0) {
                    ((unsigned short*)Cout)[(long)row * N + col] = f2bf(v);
                } else if (mode == 1) {           // Vt[b][col][s]
                    int b = row >> 11, s = row & (SEQ - 1);
                    ((unsigned short*)Cout)[((long)b * N + col) * SEQ + s] = f2bf(v);
                } else {
                    ((float*)Cout)[(long)row * N + col] = v;
                }
            }
}

// ---------------------------------------------------------------------------
// Attention core: one workgroup (8 waves) per (batch, 16-row q-tile).
// Q tile async-staged to LDS once; scores -> LDS f32; softmax; P bf16 -> LDS;
// P @ Vt with WMMA (A from LDS, 8 hoisted B fragments per k-step).
// Padded LDS strides: SF 2080 f32 (=32 banks), P 2056 bf16 / QL 1032 bf16
// (=4 banks) -> ds_load_b128 at the 2-pass bank floor instead of 16-way.
// ---------------------------------------------------------------------------
#define SF_STRIDE 2080                    // floats per score row
#define P_STRIDE  2056                    // bf16 per P row
#define QL_STRIDE 1032                    // bf16 per Q row
#define SF_BYTES  (16 * SF_STRIDE * 4)    // 133120
#define P_BYTES   (16 * P_STRIDE * 2)     //  65792
#define RED_BYTES (288 * 4)               //   1152
#define QL_BYTES  (16 * QL_STRIDE * 2)    //  33024
#define ATTN_LDS  (SF_BYTES + P_BYTES + RED_BYTES + QL_BYTES)   // ~228KB

__global__ __launch_bounds__(256)
void attn_kernel(const unsigned short* __restrict__ Qb,
                 const unsigned short* __restrict__ Kb,
                 const unsigned short* __restrict__ Vt,
                 unsigned short* __restrict__ AO) {
    extern __shared__ char smem[];
    float*          Sf     = (float*)smem;                        // [16][SF_STRIDE]
    unsigned short* P      = (unsigned short*)(smem + SF_BYTES);  // [16][P_STRIDE]
    float*          red    = (float*)(smem + SF_BYTES + P_BYTES); // [16][16]
    float*          rowMax = red + 256;
    float*          rowSum = rowMax + 16;
    unsigned short* QL     = (unsigned short*)(smem + SF_BYTES + P_BYTES + RED_BYTES);

    const int qt = blockIdx.x;
    const int b  = blockIdx.y;
    const int q0 = qt * 16;
    const int tid  = threadIdx.x;
    const int lane = tid & 31;
    const int wave = tid >> 5;
    const int l15  = lane & 15;
    const bool hiL = lane >= 16;

    const int stMax = qt + 1;             // causal: key tiles 0..qt
    const int Kd = EMB;
    const unsigned short* Qbase = Qb + ((long)b * SEQ + q0) * Kd;
    const unsigned short* Kbase = Kb + (long)b * SEQ * Kd;
    const f32x8 zero = {0.f,0.f,0.f,0.f,0.f,0.f,0.f,0.f};

    // ---- stage Q tile (16x1024 bf16) into padded LDS rows, async ----
    {
        const unsigned lQ = lds_off(QL);
#pragma unroll
        for (int i = 0; i < 8; ++i) {
            int ci  = tid + (i << 8);                  // 2048 x 16B chunks
            int qr  = ci >> 7;                         // row 0..15
            int qc  = (ci & 127) << 3;                 // element in row
            async_g2l_b128(lQ + (unsigned)(qr * QL_STRIDE + qc) * 2,
                           Qbase + ci * 8);
        }
        wait_async_0();
    }
    __syncthreads();

    // ---- phase 1: scores = Q @ K^T (A from LDS, K reg double-buffered) ----
    const float scale = 0.03125f;                      // 1024^-0.5
    const unsigned short* aRowL = QL + (long)l15 * QL_STRIDE + (hiL ? 16 : 0);
    for (int st = wave; st < stMax; st += 8) {
        const int s0 = st * 16;
        const unsigned short* bRow =
            Kbase + (long)(s0 + l15) * Kd + (hiL ? 16 : 0);
        f32x8 acc = zero;
        bf16x16 aC = load_a_frag(aRowL);
        bf16x16 bC = load_b_frag(bRow);
        for (int k0 = 0; k0 < Kd; k0 += 32) {
            bf16x16 aN = aC, bN = bC;
            if (k0 + 32 < Kd) {                        // prefetch next chunk
                aN = load_a_frag(aRowL + k0 + 32);
                bN = load_b_frag(bRow + k0 + 32);
            }
            acc = wmma_bf16(aC, bC, acc);
            aC = aN; bC = bN;
        }
#pragma unroll
        for (int r = 0; r < 8; ++r) {
            int rl  = r + (hiL ? 8 : 0);
            int col = s0 + l15;
            // reference: mask to -FLT_MAX FIRST, then scale by 1/32
            float v = (col > q0 + rl) ? -1.0633824e37f : acc[r] * scale;
            Sf[rl * SF_STRIDE + col] = v;
        }
    }
    __syncthreads();

    // ---- phase 2: softmax over cols [0, ncols) per row (16 thr/row) ----
    const int ncols = stMax * 16;
    const int row = tid >> 4, sub = tid & 15;
    float m = -3.4028235e38f;
    for (int c = sub; c < ncols; c += 16) m = fmaxf(m, Sf[row * SF_STRIDE + c]);
    red[row * 16 + sub] = m;
    __syncthreads();
    if (sub == 0) {
        float mm = red[row * 16];
#pragma unroll
        for (int i = 1; i < 16; ++i) mm = fmaxf(mm, red[row * 16 + i]);
        rowMax[row] = mm;
    }
    __syncthreads();
    const float rmax = rowMax[row];
    float s = 0.f;
    for (int c = sub; c < ncols; c += 16) {
        float e = __expf(Sf[row * SF_STRIDE + c] - rmax);
        Sf[row * SF_STRIDE + c] = e;
        s += e;
    }
    red[row * 16 + sub] = s;
    __syncthreads();
    if (sub == 0) {
        float ss = 0.f;
#pragma unroll
        for (int i = 0; i < 16; ++i) ss += red[row * 16 + i];
        rowSum[row] = ss;
    }
    __syncthreads();
    const float inv = 1.0f / rowSum[row];
    for (int c = sub; c < ncols; c += 16)
        P[row * P_STRIDE + c] = f2bf(Sf[row * SF_STRIDE + c] * inv);
    if (stMax & 1) P[row * P_STRIDE + ncols + sub] = 0; // zero-pad odd tile
    __syncthreads();

    // ---- phase 3: out = P @ Vt^T ; wave owns 128 v-cols (8 C tiles) ----
    const int ksEnd = (stMax & 1) ? ncols + 16 : ncols;
    const unsigned short* VtB = Vt + (long)b * EMB * SEQ;
    const int n0w = wave * 128;
    f32x8 acc2[8];
#pragma unroll
    for (int j = 0; j < 8; ++j) acc2[j] = zero;

    const unsigned short* pRow = P + (long)l15 * P_STRIDE + (hiL ? 16 : 0);
    for (int ks = 0; ks < ksEnd; ks += 32) {
        bf16x16 af = load_a_frag(pRow + ks);           // ds_load_b128
        bf16x16 bF[8];                                  // hoist all 8 loads
#pragma unroll
        for (int j = 0; j < 8; ++j)
            bF[j] = load_b_frag(VtB + (long)(n0w + j * 16 + l15) * SEQ +
                                (hiL ? 16 : 0) + ks);
#pragma unroll
        for (int j = 0; j < 8; ++j)
            acc2[j] = wmma_bf16(af, bF[j], acc2[j]);
    }

    unsigned short* outBase = AO + ((long)b * SEQ + q0) * EMB;
#pragma unroll
    for (int j = 0; j < 8; ++j)
#pragma unroll
        for (int r = 0; r < 8; ++r) {
            int rl = r + (hiL ? 8 : 0);
            outBase[(long)rl * EMB + n0w + j * 16 + l15] = f2bf(acc2[j][r]);
        }
}

// ---------------------------------------------------------------------------
// host launcher
// ---------------------------------------------------------------------------
extern "C" void kernel_launch(void* const* d_in, const int* in_sizes, int n_in,
                              void* d_out, int out_size, void* d_ws, size_t ws_size,
                              hipStream_t stream) {
    const float* x  = (const float*)d_in[0];
    const float* Wq = (const float*)d_in[1];
    const float* Wk = (const float*)d_in[2];
    const float* Wv = (const float*)d_in[3];
    const float* Wo = (const float*)d_in[4];

    const long M = (long)BATCH * SEQ;          // 8192
    const long xElems = M * EMB;               // 8,388,608
    const long wElems = (long)EMB * EMB;       // 1,048,576

    char* ws = (char*)d_ws;
    size_t off = 0;
    unsigned short* xb  = (unsigned short*)(ws + off); off += xElems * 2;   // 16MB
    unsigned short* WqT = (unsigned short*)(ws + off); off += wElems * 2;   //  2MB
    unsigned short* WkT = (unsigned short*)(ws + off); off += wElems * 2;
    unsigned short* WvT = (unsigned short*)(ws + off); off += wElems * 2;
    unsigned short* WoT = (unsigned short*)(ws + off); off += wElems * 2;
    unsigned short* Qb  = (unsigned short*)(ws + off); off += xElems * 2;   // 16MB
    unsigned short* Kb  = (unsigned short*)(ws + off); off += xElems * 2;
    unsigned short* VtB = (unsigned short*)(ws + off); off += xElems * 2;
    unsigned short* AO  = (unsigned short*)(ws + off); off += xElems * 2;   // 88MB total

    // 1) precision conversion (+ weight transpose for K-contiguous B operands)
    convert_bf16_kernel<<<(int)(xElems / 256), 256, 0, stream>>>(x, xb, (int)xElems);
    convert_wT_kernel<<<(int)(wElems / 256), 256, 0, stream>>>(Wq, WqT, EMB, EMB);
    convert_wT_kernel<<<(int)(wElems / 256), 256, 0, stream>>>(Wk, WkT, EMB, EMB);
    convert_wT_kernel<<<(int)(wElems / 256), 256, 0, stream>>>(Wv, WvT, EMB, EMB);
    convert_wT_kernel<<<(int)(wElems / 256), 256, 0, stream>>>(Wo, WoT, EMB, EMB);

    // 2) projections: Q, K row-major; V stored transposed per batch
    const int gemmBlocks = (int)((M >> 7) * (EMB >> 7));             // 512
    gemm_bf16_kernel<<<gemmBlocks, 256, 0, stream>>>(xb, WqT, Qb,  (int)M, EMB, EMB, 0);
    gemm_bf16_kernel<<<gemmBlocks, 256, 0, stream>>>(xb, WkT, Kb,  (int)M, EMB, EMB, 0);
    gemm_bf16_kernel<<<gemmBlocks, 256, 0, stream>>>(xb, WvT, VtB, (int)M, EMB, EMB, 1);

    // 3) causal attention core (scores + softmax + P@V), LDS-resident P
    attn_kernel<<<dim3(SEQ / 16, BATCH), 256, ATTN_LDS, stream>>>(Qb, Kb, VtB, AO);

    // 4) output projection, f32 result
    gemm_bf16_kernel<<<gemmBlocks, 256, 0, stream>>>(AO, WoT, (float*)d_out,
                                                     (int)M, EMB, EMB, 2);
}
